// linear_cqn_79937931313515
// MI455X (gfx1250) — matrix operand order repaired and checked by reading, not compiled, not run on hardware
//
#include <hip/hip_runtime.h>
#include <hip/hip_bf16.h>
#include <math.h>

// ---------------------------------------------------------------------------
// Problem constants (from the reference)
// ---------------------------------------------------------------------------
#define SB 32           // batch
#define SL 256          // inner length
#define SS 128          // state size
#define SR 4            // reward size
#define SA 16           // action size
#define SM (SB*SL)      // 8192 GEMM rows
#define D0 132          // S+R
#define D0P 192         // D0 padded to multiple of 64 (K-step of layer 1)
#define D1 2112
#define D2 4224
#define D3 8448
#define D4 4224
#define D5 64           // A*R

#define BK 64           // GEMM K-step per staged buffer (2 WMMA sub-steps)
#define LDSROW 72       // f16 per LDS B-row (144B: 16B-aligned, 36-dword stride
                        //  -> all 16 fragment lanes hit distinct banks)
#define BUFH   (64 * LDSROW)          // f16 per B staging buffer (9216B)
#define EROW   36                     // f16 per epilogue LDS row (72B)
#define SMEMH  (2 * BUFH)             // == 8 waves x 32 x EROW (epilogue reuse)

typedef __attribute__((ext_vector_type(16))) _Float16 v16h;
typedef __attribute__((ext_vector_type(8)))  _Float16 v8h;
typedef __attribute__((ext_vector_type(4)))  _Float16 v4h;
typedef __attribute__((ext_vector_type(8)))  float    v8f;

// ---------------------------------------------------------------------------
// Small helpers
// ---------------------------------------------------------------------------
__device__ __forceinline__ unsigned lds_off(const void* p)
{
    // Low 32 bits of a generic LDS pointer = group-segment byte offset
    // (ISA: flat/async LDS addr = ADDR[31:0], wave LDS_BASE added by HW).
    return (unsigned)(unsigned long long)p;
}

__device__ __forceinline__ void async_ld_b128(unsigned lds_addr, const void* gaddr)
{
    // GLOBAL_LOAD_ASYNC_TO_LDS_B128 (GV mode): vdst = LDS byte address,
    // vaddr = 64-bit global address. Tracked with ASYNCcnt.
    asm volatile("global_load_async_to_lds_b128 %0, %1, off"
                 :: "v"(lds_addr), "v"(gaddr)
                 : "memory");
}

__device__ __forceinline__ v16h frag_gap16(const _Float16* p)
{
    // A 16x32 f16 layout: lane holds 8 contiguous f16 at +0 and 8 at +16
    v8h lo = *(const v8h*)(p);
    v8h hi = *(const v8h*)(p + 16);
    return __builtin_shufflevector(lo, hi, 0,1,2,3,4,5,6,7,8,9,10,11,12,13,14,15);
}

__device__ __forceinline__ v16h frag_contig16(const _Float16* p)
{
    // Contiguous k order (unused; B here uses frag_gap16): lane holds 16 contiguous f16 (two b128 reads)
    v8h lo = *(const v8h*)(p);
    v8h hi = *(const v8h*)(p + 8);
    return __builtin_shufflevector(lo, hi, 0,1,2,3,4,5,6,7,8,9,10,11,12,13,14,15);
}

// ---------------------------------------------------------------------------
// Packing / conversion kernels
// ---------------------------------------------------------------------------

// x0[row][col] f16, [8192 x 192]: col<128 -> state, col<132 -> preference, else 0
__global__ __launch_bounds__(256) void k_pack_x0(
    const float* __restrict__ state, const float* __restrict__ pref,
    _Float16* __restrict__ x0)
{
    int idx = blockIdx.x * 256 + threadIdx.x;
    int total = SM * D0P;
    if (idx >= total) return;
    int row = idx / D0P;
    int col = idx - row * D0P;
    float v;
    if (col < SS)       v = state[(size_t)row * SS + col];
    else if (col < D0)  v = pref[(size_t)row * SR + (col - SS)];
    else                v = 0.0f;
    x0[idx] = (_Float16)v;
}

// W1t[n][k] f16, [2112 x 192]: k<132 -> W1[k*2112+n], else 0 (small, elementwise)
__global__ __launch_bounds__(256) void k_conv_w1t(
    const float* __restrict__ W1, _Float16* __restrict__ W1t)
{
    int idx = blockIdx.x * 256 + threadIdx.x;
    int total = D1 * D0P;
    if (idx >= total) return;
    int n = idx / D0P;
    int k = idx - n * D0P;
    float v = (k < D0) ? W1[(size_t)k * D1 + n] : 0.0f;
    W1t[idx] = (_Float16)v;
}

// LDS-tiled transpose + f32->f16: Wt[n][k] = W[k*N + n].
// K, N must be multiples of 32 (true for layers 2..5).
// Grid: (N/32, K/32), 256 threads (32 x 8).
__global__ __launch_bounds__(256) void k_transpose_wt(
    const float* __restrict__ W, _Float16* __restrict__ Wt, int K, int N)
{
    __shared__ float tile[32][33];
    const int tx = threadIdx.x & 31;
    const int ty = threadIdx.x >> 5;            // 0..7
    const int nb = blockIdx.x * 32;
    const int kb = blockIdx.y * 32;

    #pragma unroll
    for (int j = 0; j < 4; ++j) {               // coalesced along n
        int k = kb + ty + j * 8;
        tile[ty + j * 8][tx] = W[(size_t)k * N + (nb + tx)];
    }
    __syncthreads();
    #pragma unroll
    for (int j = 0; j < 4; ++j) {               // coalesced along k
        int n = nb + ty + j * 8;
        Wt[(size_t)n * K + (kb + tx)] = (_Float16)tile[tx][ty + j * 8];
    }
}

// ---------------------------------------------------------------------------
// WMMA GEMM:  C[M,N] = relu?( A[M,K](f16) * Bt[N,K]^T (f16) + bias[N] )
// Block: 256 threads = 8 waves (4M x 2N), tile 128(M) x 64(N); wave tile 32x32.
// B tile (64x64 f16 per K-step) staged via double-buffered async global->LDS
// DMA (2 b128 per thread per step) and shared by all 4 M-waves; A fragments
// load direct (2x reuse via L0). 8 WMMAs per barrier pair.
// MODE 1: f16 output + relu, coalesced LDS-bounce epilogue (b64 stores).
// MODE 0: f32 output, no relu (layer 5, tiny N).
// K multiple of 64; M multiple of 128; N multiple of 64.
// ---------------------------------------------------------------------------
template<int MODE>
__global__ __launch_bounds__(256) void k_gemm_wmma(
    const _Float16* __restrict__ A, const _Float16* __restrict__ Bt,
    const float* __restrict__ bias, void* __restrict__ outp,
    int N, int K)
{
    // Shared block: 2 x 9216B B-staging buffers during the K loop, then the
    // same storage is reused as the epilogue tile scratch (after barrier).
    __shared__ _Float16 smem[SMEMH];

    const int tid   = threadIdx.x;
    const int lane  = tid & 31;
    const int wave  = tid >> 5;
    const int waveM = wave & 3;
    const int waveN = wave >> 2;
    const int m_wave = blockIdx.y * 128 + waveM * 32;
    const int n0     = blockIdx.x * 64;
    const int lhalf  = lane >> 4;   // 0 or 1
    const int l15    = lane & 15;

    // Async staging: thread t copies 16 f16 of B row (n0 + t/4),
    // K-chunk (t%4)*16, as two b128 DMAs per K-step.
    const int ln  = tid >> 2;                   // 0..63
    const int lkc = tid & 3;                    // 0..3
    const _Float16* gB = Bt + (size_t)(n0 + ln) * K + lkc * 16;
    const unsigned ldsDst0 = lds_off(&smem[ln * LDSROW + lkc * 16]);
    const unsigned ldsDst1 = ldsDst0 + BUFH * 2;   // byte offset of buffer 1

    v8f acc[2][2] = {};

    // A fragment base pointers (lane-specific): row m, K-offset lhalf*8
    const _Float16* aptr0 = A + (size_t)(m_wave + l15) * K + lhalf * 8;
    const _Float16* aptr1 = aptr0 + (size_t)16 * K;
    // B fragment base offset into staging buffer: row (waveN*32 + ni*16 + l15),
    // 8 f16 at K-offset (s*32 + lhalf*8) and 8 at +16.
    const int boff = (waveN * 32 + l15) * LDSROW + lhalf * 8;

    const int nk = K / BK;

    auto issue = [&](unsigned dst, int step) {
        const _Float16* g = gB + (size_t)step * BK;
        async_ld_b128(dst,      g);
        async_ld_b128(dst + 16, g + 8);
    };
    auto compute = [&](int buf, int kbase) {
        #pragma unroll
        for (int s = 0; s < 2; ++s) {
            const int k0 = kbase + s * 32;
            v16h a0 = frag_gap16(aptr0 + k0);
            v16h a1 = frag_gap16(aptr1 + k0);
            // direct __shared__ indexing keeps addrspace(3) -> ds_load_b128
            const _Float16* bp = smem + buf * BUFH + boff + s * 32;
            v16h b0 = frag_gap16(bp);
            v16h b1 = frag_gap16(bp + 16 * LDSROW);

            acc[0][0] = __builtin_amdgcn_wmma_f32_16x16x32_f16(
                false, a0, false, b0, (short)0, acc[0][0], false, false);
            acc[0][1] = __builtin_amdgcn_wmma_f32_16x16x32_f16(
                false, a0, false, b1, (short)0, acc[0][1], false, false);
            acc[1][0] = __builtin_amdgcn_wmma_f32_16x16x32_f16(
                false, a1, false, b0, (short)0, acc[1][0], false, false);
            acc[1][1] = __builtin_amdgcn_wmma_f32_16x16x32_f16(
                false, a1, false, b1, (short)0, acc[1][1], false, false);
        }
    };

    issue(ldsDst0, 0);                          // prologue: stage K-step 0

    for (int i = 0; i < nk - 1; ++i) {          // main loop: branch-free body
        issue((i & 1) ? ldsDst0 : ldsDst1, i + 1);
        asm volatile("s_wait_asynccnt 0x2" ::: "memory");  // buf i&1 landed
        __syncthreads();
        compute(i & 1, i * BK);
        __syncthreads();                        // done reading before re-stage
    }
    asm volatile("s_wait_asynccnt 0x0" ::: "memory");      // drain
    __syncthreads();
    compute((nk - 1) & 1, (nk - 1) * BK);
    __syncthreads();                            // smem dead, safe to reuse

    if (MODE == 1) {
        // ---- f16 + relu epilogue: bounce 32x32 wave tile through LDS so
        //      global stores are b64 with 64B-contiguous lane groups.
        _Float16* wtile = smem + wave * (32 * EROW);
        #pragma unroll
        for (int ni = 0; ni < 2; ++ni) {
            float bv = bias[n0 + waveN * 32 + ni * 16 + l15];
            #pragma unroll
            for (int mi = 0; mi < 2; ++mi) {
                #pragma unroll
                for (int r = 0; r < 8; ++r) {
                    float v = acc[mi][ni][r] + bv;
                    v = fmaxf(v, 0.0f);
                    wtile[(mi * 16 + lhalf * 8 + r) * EROW + ni * 16 + l15] =
                        (_Float16)v;
                }
            }
        }
        __syncthreads();
        _Float16* outH = (_Float16*)outp;
        const int tr4 = lane >> 3;              // 0..3  (row within group)
        const int cc  = (lane & 7) * 4;         // f16 col chunk: 8 lanes x 8B
        #pragma unroll
        for (int sg = 0; sg < 8; ++sg) {
            int tr = sg * 4 + tr4;
            v4h d = *(const v4h*)&wtile[tr * EROW + cc];
            *(v4h*)&outH[(size_t)(m_wave + tr) * N + (n0 + waveN * 32 + cc)] = d;
        }
    } else {
        // ---- f32 epilogue (layer 5, 2MB total): direct stores, no relu.
        float* outF = (float*)outp;
        #pragma unroll
        for (int ni = 0; ni < 2; ++ni) {
            int col = n0 + waveN * 32 + ni * 16 + l15;
            float bv = bias[col];
            #pragma unroll
            for (int mi = 0; mi < 2; ++mi) {
                #pragma unroll
                for (int r = 0; r < 8; ++r) {
                    int row = m_wave + mi * 16 + lhalf * 8 + r;
                    outF[(size_t)row * N + col] = acc[mi][ni][r] + bv;
                }
            }
        }
    }
}

// ---------------------------------------------------------------------------
// _H epilogue. One block per batch b (32 blocks), thread = l (256 threads).
// Closed form of the reference's reshape/permute chain:
//   prod[b,aw,l] = sum_r q[(b&7)+8*(aw>>4), ((l&63)<<2)|r, aw&15, l>>6] * w[b,l,r]
//   inds = first-argmax over aw; hq rows placed by stable counting sort of inds.
// ---------------------------------------------------------------------------
__global__ __launch_bounds__(256) void k_hq(
    const float* __restrict__ pref, const float* __restrict__ q,
    float* __restrict__ hq)
{
    __shared__ int s_inds[SL];
    const int b = blockIdx.x;
    const int l = threadIdx.x;

    const float w0 = pref[((size_t)(b * SL + l)) * SR + 0];
    const float w1 = pref[((size_t)(b * SL + l)) * SR + 1];
    const float w2 = pref[((size_t)(b * SL + l)) * SR + 2];
    const float w3 = pref[((size_t)(b * SL + l)) * SR + 3];

    const int clo   = l >> 6;          // q's last-axis index
    const int lbase = (l & 63) << 2;   // base of l_orig; +r steps by 64 floats

    float best = -INFINITY;
    int   bi   = 0;
    for (int aw = 0; aw < 64; ++aw) {
        int pb = (b & 7) + 8 * (aw >> 4);
        int a  = aw & 15;
        const float* qb = q + (((size_t)pb * SL + lbase) * SA + a) * SR + clo;
        float dot = qb[0] * w0 + qb[64] * w1 + qb[128] * w2 + qb[192] * w3;
        if (dot > best) { best = dot; bi = aw; }
    }
    s_inds[l] = bi;
    __syncthreads();

    // stable rank: (#keys < mine) + (#equal keys from earlier l)
    int pos = 0;
    for (int t = 0; t < SL; ++t) {
        int k = s_inds[t];
        pos += (k < bi) | ((k == bi) & (t < l));
    }

    int pb = (b & 7) + 8 * (bi >> 4);
    int a  = bi & 15;
    const float* qb = q + (((size_t)pb * SL + lbase) * SA + a) * SR + clo;
    float* o = hq + ((size_t)b * SL + pos) * SR;
    o[0] = qb[0];
    o[1] = qb[64];
    o[2] = qb[128];
    o[3] = qb[192];
}

// ---------------------------------------------------------------------------
// Host launch
// ---------------------------------------------------------------------------
static inline size_t align256(size_t x) { return (x + 255) & ~(size_t)255; }

extern "C" void kernel_launch(void* const* d_in, const int* in_sizes, int n_in,
                              void* d_out, int out_size, void* d_ws, size_t ws_size,
                              hipStream_t stream)
{
    const float* state = (const float*)d_in[0];
    const float* pref  = (const float*)d_in[1];
    // d_in[2] = w_num (==4, baked into the index math)
    const float* W1 = (const float*)d_in[3];
    const float* b1 = (const float*)d_in[4];
    const float* W2 = (const float*)d_in[5];
    const float* b2 = (const float*)d_in[6];
    const float* W3 = (const float*)d_in[7];
    const float* b3 = (const float*)d_in[8];
    const float* W4 = (const float*)d_in[9];
    const float* b4 = (const float*)d_in[10];
    const float* W5 = (const float*)d_in[11];
    const float* b5 = (const float*)d_in[12];

    float* hq_out = (float*)d_out;                 // [8192, 4]
    float* q_out  = (float*)d_out + SM * SR;       // [32,256,16,4]

    // Workspace carve-up (f16 buffers)
    char* ws = (char*)d_ws;
    size_t off = 0;
    auto take = [&](size_t bytes) { char* p = ws + off; off += align256(bytes); return p; };
    _Float16* x0  = (_Float16*)take((size_t)SM * D0P * 2);
    _Float16* x1  = (_Float16*)take((size_t)SM * D1  * 2);
    _Float16* x2  = (_Float16*)take((size_t)SM * D2  * 2);
    _Float16* x3  = (_Float16*)take((size_t)SM * D3  * 2);
    _Float16* x4  = (_Float16*)take((size_t)SM * D4  * 2);
    _Float16* W1t = (_Float16*)take((size_t)D1 * D0P * 2);
    _Float16* W2t = (_Float16*)take((size_t)D2 * D1  * 2);
    _Float16* W3t = (_Float16*)take((size_t)D3 * D2  * 2);
    _Float16* W4t = (_Float16*)take((size_t)D4 * D3  * 2);
    _Float16* W5t = (_Float16*)take((size_t)D5 * D4  * 2);

    auto blocks = [](long long total) { return dim3((unsigned)((total + 255) / 256)); };

    // Pack / convert (LDS-tiled transposes: coalesced reads AND writes)
    k_pack_x0 <<<blocks((long long)SM * D0P), 256, 0, stream>>>(state, pref, x0);
    k_conv_w1t<<<blocks((long long)D1 * D0P), 256, 0, stream>>>(W1, W1t);
    k_transpose_wt<<<dim3(D2 / 32, D1 / 32), 256, 0, stream>>>(W2, W2t, D1, D2);
    k_transpose_wt<<<dim3(D3 / 32, D2 / 32), 256, 0, stream>>>(W3, W3t, D2, D3);
    k_transpose_wt<<<dim3(D4 / 32, D3 / 32), 256, 0, stream>>>(W4, W4t, D3, D4);
    k_transpose_wt<<<dim3(D5 / 32, D4 / 32), 256, 0, stream>>>(W5, W5t, D4, D5);

    // 5 GEMM layers (grid: N/64 x M/128)
    dim3 blk(256);
    k_gemm_wmma<1><<<dim3(D1 / 64, SM / 128), blk, 0, stream>>>(x0, W1t, b1, x1, D1, D0P);
    k_gemm_wmma<1><<<dim3(D2 / 64, SM / 128), blk, 0, stream>>>(x1, W2t, b2, x2, D2, D1);
    k_gemm_wmma<1><<<dim3(D3 / 64, SM / 128), blk, 0, stream>>>(x2, W3t, b3, x3, D3, D2);
    k_gemm_wmma<1><<<dim3(D4 / 64, SM / 128), blk, 0, stream>>>(x3, W4t, b4, x4, D4, D3);
    k_gemm_wmma<0><<<dim3(D5 / 64, SM / 128), blk, 0, stream>>>(x4, W5t, b5, q_out, D5, D4);

    // _H epilogue
    k_hq<<<dim3(SB), dim3(SL), 0, stream>>>(pref, q_out, hq_out);
}
